// Correlation_13048110645294
// MI455X (gfx1250) — matrix operand order repaired and checked by reading, not compiled, not run on hardware
//
#include <hip/hip_runtime.h>

#define B_N   32
#define H_N   96
#define W_N   96
#define C_N   128                     // C_IN == C_FEAT
#define HW_N  (H_N * W_N)             // 9216
#define NPIX  ((size_t)B_N * HW_N)    // 294912 pixels per image
#define PSTR  136                     // padded pixel record stride (bf16 elems, 272B = 17*16B)
#define SCALE 0.08838834764831845f    // 1/sqrt(128)

typedef __attribute__((ext_vector_type(16))) __bf16 v16bf;
typedef __attribute__((ext_vector_type(2)))  __bf16 v2bf;
typedef __attribute__((ext_vector_type(8)))  float  v8f;

struct __align__(16) U4 { unsigned x, y, z, w; };
union BFV { v16bf v; U4 q[2]; };

// Pack two f32 -> packed bf16x2 via native bf16 casts (RNE); lowers to the
// hardware packed convert on gfx1250 instead of a manual bit-twiddle sequence.
__device__ __forceinline__ unsigned pk2(float a, float b) {
  union { v2bf h; unsigned u; } c;
  c.h = (v2bf){(__bf16)a, (__bf16)b};
  return c.u;
}

// ---------------------------------------------------------------------------
// Kernel 1: per-pixel projection  x[f,p] = sum_c W[f,c]*in[c,p] + bias[f]
// Output layout: bf16, pixel-major [b][h][w][c] (256B per pixel record).
// ---------------------------------------------------------------------------
#define PROJ_SMEM (3 * 128 * PSTR * 2 + 512)

__global__ void __launch_bounds__(256)
proj_kernel(const float* __restrict__ in1, const float* __restrict__ in2,
            const float* __restrict__ pw,  const float* __restrict__ pb,
            unsigned short* __restrict__ x1ws, unsigned short* __restrict__ x2ws)
{
  extern __shared__ __align__(16) char smem[];
  unsigned short* Wl   = (unsigned short*)smem;        // [128 f][PSTR]
  unsigned short* Xl   = Wl + 128 * PSTR;              // [128 px][PSTR]
  unsigned short* OUTl = Xl + 128 * PSTR;              // [128 px][PSTR]
  float*          biasl = (float*)(OUTl + 128 * PSTR); // [128]

  const int tid = threadIdx.x;
  const int b   = blockIdx.y;
  const int p0  = blockIdx.x * 128;
  const float* src = blockIdx.z ? in2 : in1;
  unsigned short* dst = blockIdx.z ? x2ws : x1ws;

  // Stage W: f32 [f][c] -> bf16 LDS [f][c] (padded rows)
  for (int i = tid; i < 128 * 64; i += 256) {
    int f = i >> 6, cp = (i & 63) * 2;
    const float* g = pw + f * 128 + cp;
    *(unsigned*)(Wl + f * PSTR + cp) = pk2(g[0], g[1]);
  }
  if (tid < 128) biasl[tid] = pb[tid];
  // Stage X: global [c][p] -> LDS [px][c] transpose (bf16 pairs)
  for (int i = tid; i < 128 * 64; i += 256) {
    int px = i & 127, c = (i >> 7) * 2;
    const float* g = src + ((size_t)b * C_N + c) * HW_N + p0 + px;
    *(unsigned*)(Xl + px * PSTR + c) = pk2(g[0], g[HW_N]);
  }
  __syncthreads();

  const int lane = tid & 31, wv = tid >> 5;
  const int ln = lane & 15, hi = lane >> 4;
  const int wp0 = wv * 16;

  // Preload B operand (16 pixels x K=128 split into 4 K-tiles)
  BFV bt[4];
#pragma unroll
  for (int k = 0; k < 4; ++k) {
    const unsigned short* xp = Xl + (wp0 + ln) * PSTR + k * 32 + hi * 16;
    bt[k].q[0] = *(const U4*)xp;
    bt[k].q[1] = *(const U4*)(xp + 8);
  }

#pragma unroll
  for (int mt = 0; mt < 8; ++mt) {
    v8f acc = {0.f, 0.f, 0.f, 0.f, 0.f, 0.f, 0.f, 0.f};
#pragma unroll
    for (int k = 0; k < 4; ++k) {
      BFV a;  // A operand: W rows mt*16..mt*16+15, contiguous K order k = 16*hi + s, same as B
      const unsigned short* wp = Wl + (mt * 16 + ln) * PSTR + k * 32 + hi * 16;
      a.q[0] = *(const U4*)wp;
      a.q[1] = *(const U4*)(wp + 8);
      acc = __builtin_amdgcn_wmma_f32_16x16x32_bf16(false, a.v, false, bt[k].v,
                                                    (short)0, acc, false, false);
    }
    const float* bp = biasl + mt * 16 + hi * 8;   // f = mt*16 + 8*hi + r
    U4 o;
    o.x = pk2(acc[0] + bp[0], acc[1] + bp[1]);
    o.y = pk2(acc[2] + bp[2], acc[3] + bp[3]);
    o.z = pk2(acc[4] + bp[4], acc[5] + bp[5]);
    o.w = pk2(acc[6] + bp[6], acc[7] + bp[7]);
    *(U4*)(OUTl + (wp0 + ln) * PSTR + mt * 16 + hi * 8) = o;
  }

  // Coalesced copy of this wave's 16px x 128c bf16 tile to global [b][p][c]
#pragma unroll
  for (int i = 0; i < 8; ++i) {
    int chunk = i * 32 + lane;                   // 256 x 16B chunks
    int px = chunk >> 4, off = (chunk & 15) * 8; // off in ushorts
    U4 d = *(const U4*)(OUTl + (wp0 + px) * PSTR + off);
    *(U4*)(dst + ((size_t)b * HW_N + p0 + wp0 + px) * C_N + off) = d;
  }
}

// ---------------------------------------------------------------------------
// Kernel 2: 9x9 local correlation via banded all-pairs row GEMM.
// Workgroup: (b, 8 rows, 16 cols). Wave r handles row h0+r:
//   Corr[w,u] = sum_c x1[c,w]*x2[c,u]  over 32-wide u window -> 8 WMMAs per di.
//   out[di,dj,h,w] = Corr[w, w+dj-4] / sqrt(128)
// ---------------------------------------------------------------------------
#define X2ROW (32 * PSTR)
#define X1ROW (16 * PSTR)
#define CORR_SMEM (16 * X2ROW * 2 + 8 * X1ROW * 2 + 8 * 528 * 4)

__global__ void __launch_bounds__(256)
corr_kernel(const unsigned short* __restrict__ x1ws,
            const unsigned short* __restrict__ x2ws,
            float* __restrict__ out)
{
  extern __shared__ __align__(16) char smem[];
  unsigned short* x2l = (unsigned short*)smem;       // [16 rows][32 px][PSTR]
  unsigned short* x1l = x2l + 16 * X2ROW;            // [8 rows][16 px][PSTR]
  float*          corr = (float*)(x1l + 8 * X1ROW);  // [8 waves][16][33]

  const int tid = threadIdx.x;
  const int lane = tid & 31, wv = tid >> 5;
  const int ln = lane & 15, hi = lane >> 4;
  const int b  = blockIdx.z;
  const int h0 = blockIdx.y * 8;
  const int w0 = blockIdx.x * 16;

  // Stage x2 halo tile: rows h0-4..h0+11, cols w0-4..w0+27, zero-padded OOB
  for (int seg = wv; seg < 16; seg += 8) {
    int hr = h0 - 4 + seg;
    bool hok = (unsigned)hr < (unsigned)H_N;
#pragma unroll 4
    for (int i = 0; i < 16; ++i) {
      int chunk = i * 32 + lane;                   // 512 x 16B chunks
      int px = chunk >> 4;                         // 0..31
      int off = (chunk & 15) * 8;                  // ushort offset in pixel
      U4 d; d.x = d.y = d.z = d.w = 0u;
      if (hok) {
        int wj = w0 - 4 + px;
        if ((unsigned)wj < (unsigned)W_N)
          d = *(const U4*)(x2ws + (((size_t)b * H_N + hr) * W_N + wj) * C_N + off);
      }
      *(U4*)(x2l + seg * X2ROW + px * PSTR + off) = d;
    }
  }
  // Stage x1 tile: rows h0..h0+7, cols w0..w0+15 (always in-bounds)
  {
    const unsigned short* rowg = x1ws + (((size_t)b * H_N + h0 + wv) * W_N + w0) * C_N;
#pragma unroll
    for (int i = 0; i < 8; ++i) {
      int chunk = i * 32 + lane;
      int px = chunk >> 4, off = (chunk & 15) * 8;
      U4 d = *(const U4*)(rowg + (size_t)px * C_N + off);
      *(U4*)(x1l + wv * X1ROW + px * PSTR + off) = d;
    }
  }
  __syncthreads();

  // Preload A operand (x1 pixels of this wave's row), 4 K-tiles (stay resident)
  BFV A[4];
#pragma unroll
  for (int k = 0; k < 4; ++k) {
    const unsigned short* p = x1l + wv * X1ROW + ln * PSTR + k * 32 + hi * 16;
    A[k].q[0] = *(const U4*)p;
    A[k].q[1] = *(const U4*)(p + 8);
  }
  float* cw = corr + wv * 528;           // 16 x 33 (padded) scratch
  const int h = h0 + wv;

  // Fully unrolled over di so the scheduler can overlap iteration di+1's
  // ds_load_b128 + WMMA stream with iteration di's extraction/stores.
#pragma unroll
  for (int di = 0; di < 9; ++di) {
    const unsigned short* xr = x2l + (wv + di) * X2ROW;
    v8f acc0 = {0.f, 0.f, 0.f, 0.f, 0.f, 0.f, 0.f, 0.f};
    v8f acc1 = {0.f, 0.f, 0.f, 0.f, 0.f, 0.f, 0.f, 0.f};
#pragma unroll
    for (int k = 0; k < 4; ++k) {
      BFV b0, b1;  // B operand: lanes<16 K 0..15, lanes>=16 K 16..31
      const unsigned short* q0 = xr + ln * PSTR + k * 32 + hi * 16;
      b0.q[0] = *(const U4*)q0;
      b0.q[1] = *(const U4*)(q0 + 8);
      const unsigned short* q1 = q0 + 16 * PSTR;
      b1.q[0] = *(const U4*)q1;
      b1.q[1] = *(const U4*)(q1 + 8);
      acc0 = __builtin_amdgcn_wmma_f32_16x16x32_bf16(false, A[k].v, false, b0.v,
                                                     (short)0, acc0, false, false);
      acc1 = __builtin_amdgcn_wmma_f32_16x16x32_bf16(false, A[k].v, false, b1.v,
                                                     (short)0, acc1, false, false);
    }
    // Dump 16x32 Corr tile to padded LDS scratch (D layout: lane=N, VGPR r = M)
#pragma unroll
    for (int r = 0; r < 8; ++r) {
      int m = r + hi * 8;
      cw[m * 33 + ln]      = acc0[r];
      cw[m * 33 + 16 + ln] = acc1[r];
    }
    // Extract the 9 diagonals: out[dj, w=ln] = Corr[ln, ln+dj]
#pragma unroll
    for (int djb = 0; djb < 9; djb += 2) {
      int dj = djb + hi;
      if (dj < 9) {
        float v = cw[ln * 34 + dj] * SCALE;
        out[(((size_t)b * 81 + di * 9 + dj) * H_N + h) * W_N + w0 + ln] = v;
      }
    }
  }
}

// ---------------------------------------------------------------------------
extern "C" void kernel_launch(void* const* d_in, const int* in_sizes, int n_in,
                              void* d_out, int out_size, void* d_ws, size_t ws_size,
                              hipStream_t stream)
{
  (void)in_sizes; (void)n_in; (void)out_size; (void)ws_size;
  const float* in1 = (const float*)d_in[0];
  const float* in2 = (const float*)d_in[1];
  const float* pw  = (const float*)d_in[2];
  const float* pb  = (const float*)d_in[3];
  float* out = (float*)d_out;

  unsigned short* x1 = (unsigned short*)d_ws;           // bf16 [b][h][w][c]
  unsigned short* x2 = x1 + NPIX * C_N;                 // bf16 [b][h][w][c]

  (void)hipFuncSetAttribute(reinterpret_cast<const void*>(proj_kernel),
                            hipFuncAttributeMaxDynamicSharedMemorySize, PROJ_SMEM);
  (void)hipFuncSetAttribute(reinterpret_cast<const void*>(corr_kernel),
                            hipFuncAttributeMaxDynamicSharedMemorySize, CORR_SMEM);

  proj_kernel<<<dim3(HW_N / 128, B_N, 2), dim3(256), PROJ_SMEM, stream>>>(
      in1, in2, pw, pb, x1, x2);
  corr_kernel<<<dim3(W_N / 16, H_N / 8, B_N), dim3(256), CORR_SMEM, stream>>>(
      x1, x2, out);
}
